// StandGATX_9122510537108
// MI455X (gfx1250) — matrix operand order repaired and checked by reading, not compiled, not run on hardware
//
#include <hip/hip_runtime.h>
#include <hip/hip_bf16.h>

typedef __attribute__((ext_vector_type(16))) __bf16   v16bf;
typedef __attribute__((ext_vector_type(8)))  float    v8f;
typedef __attribute__((ext_vector_type(4)))  unsigned uq4;

union FragBF {
    v16bf v;
    uq4   q[2];
};

// ---------- helpers ----------
__device__ __forceinline__ unsigned short f2bf(float f) {
    unsigned u = __float_as_uint(f);
    u = u + 0x7FFFu + ((u >> 16) & 1u);   // round-to-nearest-even
    return (unsigned short)(u >> 16);
}

// order-preserving float <-> uint encoding for atomicMax-based segment max
__device__ __forceinline__ unsigned encf(float f) {
    unsigned u = __float_as_uint(f);
    return (u & 0x80000000u) ? ~u : (u | 0x80000000u);
}
__device__ __forceinline__ float decf(unsigned e) {
    unsigned u = (e & 0x80000000u) ? (e & 0x7FFFFFFFu) : ~e;
    return __uint_as_float(u);
}
#define ENC_NEG_INF 0x007FFFFFu   // encf(-inf)

__device__ __forceinline__ float lrelu(float v) {
    return v > 0.0f ? v : 0.2f * v;
}

__device__ __forceinline__ void edge_pair(const int* __restrict__ adj, int E, int e,
                                          int& s, int& d) {
    if (e < E) { s = adj[e]; d = adj[E + e]; }
    else       { s = d = e - E; }            // appended self loops
}

// ---------- GEMM: H[M,128] = (reluIn? relu(X) : X)[M,128] @ W[128,128], bf16 WMMA ----------
__global__ __launch_bounds__(256)
void gat_gemm_wmma(const float* __restrict__ X, const float* __restrict__ W,
                   float* __restrict__ H, int M, int reluIn) {
    __shared__ __align__(16) unsigned short ldsX[64 * 128];    // 16 KB, row-major [m][k]
    __shared__ __align__(16) unsigned short ldsWt[128 * 128];  // 32 KB, transposed  [n][k]

    const int tid  = threadIdx.x;
    const int rowb = blockIdx.x * 64;

    // stage W transposed as bf16 (reused for all K-steps)
    for (int i = tid; i < 128 * 128; i += 256) {
        int k = i >> 7, n = i & 127;
        ldsWt[n * 128 + k] = f2bf(W[i]);
    }
    // stage X tile as bf16 (optional ReLU on load = inter-layer activation fusion)
    for (int i = tid; i < 64 * 128; i += 256) {
        int r = i >> 7;
        int row = rowb + r;
        float v = (row < M) ? X[(size_t)row * 128 + (i & 127)] : 0.0f;
        if (reluIn) v = fmaxf(v, 0.0f);
        ldsX[i] = f2bf(v);
    }
    __syncthreads();

    const int wave = tid >> 5;         // 0..7
    const int lane = tid & 31;
    const int wrow = wave >> 1;        // 4 row-tiles of 16
    const int wcol = wave & 1;         // 2 col-strips of 64
    const int half = lane >> 4;
    const int l15  = lane & 15;
    const int mloc = wrow * 16 + l15;  // A-fragment row for this lane

    v8f acc[4];
    v8f zero = {0.f, 0.f, 0.f, 0.f, 0.f, 0.f, 0.f, 0.f};
    for (int t = 0; t < 4; ++t) acc[t] = zero;

    for (int k0 = 0; k0 < 128; k0 += 32) {
        // A fragment (ISA 16-bit A layout): lane holds row mloc,
        // elems 0..7 -> K = k0+8*half+i ; elems 8..15 -> K = k0+16+8*half+i
        FragBF a;
        a.q[0] = *(const uq4*)&ldsX[mloc * 128 + k0 + 8 * half];
        a.q[1] = *(const uq4*)&ldsX[mloc * 128 + k0 + 16 + 8 * half];
        for (int t = 0; t < 4; ++t) {
            // B fragment from transposed LDS: lane holds column n,
            // elems 0..7 -> K = k0+8*half+i ; elems 8..15 -> K = k0+16+8*half+i
            int n = wcol * 64 + t * 16 + l15;
            FragBF b;
            const uq4* pb = (const uq4*)&ldsWt[n * 128 + k0 + 8 * half];
            b.q[0] = pb[0];
            b.q[1] = pb[2];
            acc[t] = __builtin_amdgcn_wmma_f32_16x16x32_bf16(
                false, a.v, false, b.v, (short)0, acc[t], false, false);
        }
    }

    // C/D layout: lane -> col n, VGPR r -> row 8*half + r
    for (int t = 0; t < 4; ++t) {
        int n = wcol * 64 + t * 16 + l15;
        for (int r = 0; r < 8; ++r) {
            int row = rowb + wrow * 16 + 8 * half + r;
            if (row < M) H[(size_t)row * 128 + n] = acc[t][r];
        }
    }
}

// ---------- attention coefficients: as/ad [M,4] ----------
__global__ void gat_attn(const float* __restrict__ H,
                         const float* __restrict__ att_s, const float* __restrict__ att_d,
                         float* __restrict__ asv, float* __restrict__ adv, int M) {
    int idx = blockIdx.x * blockDim.x + threadIdx.x;   // (node, head)
    if (idx >= M * 4) return;
    int n = idx >> 2, h = idx & 3;
    const float* hp = H + (size_t)n * 128 + h * 32;
    const float* s  = att_s + h * 32;
    const float* d  = att_d + h * 32;
    float ss = 0.f, dd = 0.f;
#pragma unroll
    for (int c = 0; c < 32; ++c) { float v = hp[c]; ss += v * s[c]; dd += v * d[c]; }
    asv[idx] = ss;
    adv[idx] = dd;
}

// ---------- per-node init: agg = bias (broadcast), m = -inf, denom = 0 ----------
__global__ void gat_init(float* __restrict__ agg, const float* __restrict__ bias,
                         unsigned* __restrict__ mEnc, float* __restrict__ denom, int M) {
    int idx = blockIdx.x * blockDim.x + threadIdx.x;
    if (idx >= M * 128) return;
    int j = idx & 127;
    agg[idx] = bias[j];
    if (j < 4) {
        int n = idx >> 7;
        mEnc[n * 4 + j]  = ENC_NEG_INF;
        denom[n * 4 + j] = 0.0f;
    }
}

// ---------- segment max over edges (order-preserving uint atomicMax) ----------
__global__ void gat_edge_max(const int* __restrict__ adj, int E, int M,
                             const float* __restrict__ asv, const float* __restrict__ adv,
                             unsigned* __restrict__ mEnc) {
    int e = blockIdx.x * blockDim.x + threadIdx.x;
    if (e >= E + M) return;
    int s, d; edge_pair(adj, E, e, s, d);
#pragma unroll
    for (int h = 0; h < 4; ++h) {
        float ev = lrelu(asv[s * 4 + h] + adv[d * 4 + h]);
        atomicMax(&mEnc[d * 4 + h], encf(ev));
    }
}

// ---------- segment exp-sum ----------
__global__ void gat_edge_sum(const int* __restrict__ adj, int E, int M,
                             const float* __restrict__ asv, const float* __restrict__ adv,
                             const unsigned* __restrict__ mEnc, float* __restrict__ denom) {
    int e = blockIdx.x * blockDim.x + threadIdx.x;
    if (e >= E + M) return;
    int s, d; edge_pair(adj, E, e, s, d);
#pragma unroll
    for (int h = 0; h < 4; ++h) {
        float ev = lrelu(asv[s * 4 + h] + adv[d * 4 + h]);
        float ex = __expf(ev - decf(mEnc[d * 4 + h]));
        atomicAdd(&denom[d * 4 + h], ex);
    }
}

// ---------- weighted scatter: one 32-lane group per (edge, head), lane = channel ----------
__global__ void gat_edge_agg(const int* __restrict__ adj, int E, int M,
                             const float* __restrict__ asv, const float* __restrict__ adv,
                             const unsigned* __restrict__ mEnc, const float* __restrict__ denom,
                             const float* __restrict__ H, float* __restrict__ agg) {
    int gid  = blockIdx.x * 8 + (threadIdx.x >> 5);   // (edge, head) id
    int lane = threadIdx.x & 31;                      // channel within head
    if (gid >= (E + M) * 4) return;
    int e = gid >> 2, h = gid & 3;
    int s, d; edge_pair(adj, E, e, s, d);
    float ev    = lrelu(asv[s * 4 + h] + adv[d * 4 + h]);
    float alpha = __expf(ev - decf(mEnc[d * 4 + h])) / (denom[d * 4 + h] + 1e-16f);
    float contrib = H[(size_t)s * 128 + h * 32 + lane] * alpha;
    atomicAdd(&agg[(size_t)d * 128 + h * 32 + lane], contrib);
}

extern "C" void kernel_launch(void* const* d_in, const int* in_sizes, int n_in,
                              void* d_out, int out_size, void* d_ws, size_t ws_size,
                              hipStream_t stream) {
    const float* x   = (const float*)d_in[0];
    const int*   adj = (const int*)d_in[1];
    const float* W1  = (const float*)d_in[2];
    const float* s1  = (const float*)d_in[3];
    const float* d1  = (const float*)d_in[4];
    const float* b1  = (const float*)d_in[5];
    const float* W2  = (const float*)d_in[6];
    const float* s2  = (const float*)d_in[7];
    const float* d2  = (const float*)d_in[8];
    const float* b2  = (const float*)d_in[9];
    float* out = (float*)d_out;

    const int M = in_sizes[0] / 128;   // 50000 nodes
    const int E = in_sizes[1] / 2;     // 800000 edges (self loops appended in-kernel)

    // workspace carve-out
    char* ws = (char*)d_ws;
    size_t featB = (size_t)M * 128 * sizeof(float);
    float*    hfeat = (float*)(ws);                 // transformed features (reused per layer)
    float*    agg1  = (float*)(ws + featB);         // layer-1 aggregate / layer-2 input
    float*    asv   = (float*)(ws + 2 * featB);     // [M,4]
    float*    adv   = asv + (size_t)M * 4;
    unsigned* mEnc  = (unsigned*)(adv + (size_t)M * 4);
    float*    denom = (float*)(mEnc + (size_t)M * 4);

    const int gGemm = (M + 63) / 64;
    const int gNH   = (M * 4 + 255) / 256;
    const int gInit = (M * 128 + 255) / 256;
    const int E2    = E + M;
    const int gEdge = (E2 + 255) / 256;
    const int gAgg  = (E2 * 4 + 7) / 8;

    // ---- layer 1 ----
    gat_gemm_wmma<<<gGemm, 256, 0, stream>>>(x, W1, hfeat, M, 0);
    gat_attn<<<gNH, 256, 0, stream>>>(hfeat, s1, d1, asv, adv, M);
    gat_init<<<gInit, 256, 0, stream>>>(agg1, b1, mEnc, denom, M);
    gat_edge_max<<<gEdge, 256, 0, stream>>>(adj, E, M, asv, adv, mEnc);
    gat_edge_sum<<<gEdge, 256, 0, stream>>>(adj, E, M, asv, adv, mEnc, denom);
    gat_edge_agg<<<gAgg, 256, 0, stream>>>(adj, E, M, asv, adv, mEnc, denom, hfeat, agg1);

    // ---- layer 2 (ReLU fused into GEMM input load; bias fused into init of d_out) ----
    gat_gemm_wmma<<<gGemm, 256, 0, stream>>>(agg1, W2, hfeat, M, 1);
    gat_attn<<<gNH, 256, 0, stream>>>(hfeat, s2, d2, asv, adv, M);
    gat_init<<<gInit, 256, 0, stream>>>(out, b2, mEnc, denom, M);
    gat_edge_max<<<gEdge, 256, 0, stream>>>(adj, E, M, asv, adv, mEnc);
    gat_edge_sum<<<gEdge, 256, 0, stream>>>(adj, E, M, asv, adv, mEnc, denom);
    gat_edge_agg<<<gAgg, 256, 0, stream>>>(adj, E, M, asv, adv, mEnc, denom, hfeat, out);
}
